// MultiHeadSelfAttention_89300960019214
// MI455X (gfx1250) — matrix operand order repaired and checked by reading, not compiled, not run on hardware
//
#include <hip/hip_runtime.h>

// ---------------------------------------------------------------------------
// MHA for MI455X (gfx1250, wave32). All GEMMs via v_wmma_f32_16x16x32_bf16.
// Round 3: explicit ping-pong double buffering in every WMMA loop so fragment
// loads for step k+1 are in flight while step k's WMMAs issue (avoids the
// s_wait_loadcnt 0x0 full-drain before each WMMA seen in round 2 asm).
// Layouts used here (A/B use contiguous k, not the ISA 7.12.2 table order):
//   A (16x32 bf16): lane half h, slot e(0..15): k = 16h+e   (row m = lane&15)
//   B (32x16 bf16): lane half h, slot e(0..15): k = 16h+e   (col n = lane&15)
//   C/D (16x16 f32): vgpr g, lane l: m = g + 8*(l>>4), n = l&15
// ---------------------------------------------------------------------------

typedef __attribute__((ext_vector_type(16))) short  v16s;
typedef __attribute__((ext_vector_type(16))) __bf16 v16bf;
typedef __attribute__((ext_vector_type(8)))  float  v8f;
typedef __attribute__((ext_vector_type(4)))  float  v4f;
typedef __attribute__((ext_vector_type(4)))  unsigned int v4u;

union Frag16 {
    v16s           s;
    unsigned short us[16];
    v4u            q[2];
};

__device__ __forceinline__ unsigned short f2bf(float f) {
    unsigned u = __builtin_bit_cast(unsigned, f);
    u += 0x7FFFu + ((u >> 16) & 1u);          // round-to-nearest-even
    return (unsigned short)(u >> 16);
}

__device__ __forceinline__ v8f wmma_bf16(const Frag16& a, const Frag16& b, v8f c) {
    return __builtin_amdgcn_wmma_f32_16x16x32_bf16(
        false, __builtin_bit_cast(v16bf, a.s),
        false, __builtin_bit_cast(v16bf, b.s),
        (short)0, c, false, false);
}

#define DMODEL 768
#define SEQ    2048
#define NHEAD  12
#define DK     64

// ---------------------------------------------------------------------------
// Kernel 0: bulk fp32 -> bf16 (8 elements / thread, b128 in, b128 out).
// ---------------------------------------------------------------------------
__global__ __launch_bounds__(256) void cvt_kernel(const float* __restrict__ in,
                                                  unsigned short* __restrict__ out,
                                                  int n8) {
    const int i = blockIdx.x * 256 + threadIdx.x;
    if (i >= n8) return;
    const v4f* p = (const v4f*)(in + (size_t)i * 8);
    v4f a = p[0], b = p[1];
    Frag16 f;   // use first 8 slots
    #pragma unroll
    for (int e = 0; e < 4; ++e) { f.us[e] = f2bf(a[e]); f.us[4 + e] = f2bf(b[e]); }
    *(v4u*)(out + (size_t)i * 8) = f.q[0];
}

// ---------------------------------------------------------------------------
// Kernel 1: q/k/v = x @ W^T + b. One wave -> 16 rows x 64 cols of one matrix.
// q,k stored bf16 [B,H,S,DK]; v stored transposed bf16 [B,H,DK,S] so the
// attention kernel's B-operands are always contiguous.
// ---------------------------------------------------------------------------
__global__ __launch_bounds__(128) void qkv_kernel(
    const unsigned short* __restrict__ xb,
    const unsigned short* __restrict__ wqb, const float* __restrict__ bq,
    const unsigned short* __restrict__ wkb, const float* __restrict__ bk,
    const unsigned short* __restrict__ wvb, const float* __restrict__ bv,
    unsigned short* __restrict__ qb, unsigned short* __restrict__ kb,
    unsigned short* __restrict__ vtb)
{
    const int wid  = blockIdx.x * 4 + (threadIdx.x >> 5);
    const int lane = threadIdx.x & 31;
    const int h4   = lane >> 4;     // lane half
    const int nl   = lane & 15;

    const int mat      = wid / 3072;           // 0:q 1:k 2:v
    const int rem      = wid - mat * 3072;
    const int rowTile  = rem / 12;             // 0..255  (16 tokens each)
    const int colGroup = rem % 12;             // 0..11   (64 features each)

    const unsigned short* W = (mat == 0) ? wqb : (mat == 1) ? wkb : wvb;
    const float* bias       = (mat == 0) ? bq  : (mat == 1) ? bk  : bv;

    const unsigned short* xrow = xb + (size_t)(rowTile * 16 + nl) * DMODEL;
    const unsigned short* wbase = W + (size_t)(colGroup * 64 + nl) * DMODEL + 16 * h4;

    auto loadA = [&](Frag16& a, int k0) {
        a.q[0] = *(const v4u*)(xrow + k0 + 16 * h4);
        a.q[1] = *(const v4u*)(xrow + k0 + 16 * h4 + 8);
    };
    auto loadB = [&](Frag16* bf, int k0) {
        #pragma unroll
        for (int t = 0; t < 4; ++t) {
            const unsigned short* wr = wbase + (size_t)t * 16 * DMODEL + k0;
            bf[t].q[0] = *(const v4u*)(wr);
            bf[t].q[1] = *(const v4u*)(wr + 8);
        }
    };

    v8f acc[4] = {};
    Frag16 a0, a1, b0[4], b1[4];
    loadA(a0, 0); loadB(b0, 0);

    for (int k0 = 0; k0 < DMODEL; k0 += 64) {
        loadA(a1, k0 + 32); loadB(b1, k0 + 32);        // k0+32 <= 736, in range
        #pragma unroll
        for (int t = 0; t < 4; ++t) acc[t] = wmma_bf16(a0, b0[t], acc[t]);
        const int kn = (k0 + 64 < DMODEL) ? k0 + 64 : 0;  // safe wrap on tail
        loadA(a0, kn); loadB(b0, kn);
        #pragma unroll
        for (int t = 0; t < 4; ++t) acc[t] = wmma_bf16(a1, b1[t], acc[t]);
    }

    #pragma unroll
    for (int t = 0; t < 4; ++t) {
        const int d    = colGroup * 64 + t * 16 + nl;
        const float bb = bias[d];
        const int head = d >> 6, dk = d & 63;
        #pragma unroll
        for (int g = 0; g < 8; ++g) {
            const int m = g + 8 * h4;
            const int r = rowTile * 16 + m;        // global token
            const int bi = r >> 11, s = r & 2047;
            const unsigned short v = f2bf(acc[t][g] + bb);
            if (mat == 0)      qb [(((size_t)bi * NHEAD + head) * SEQ + s) * DK + dk] = v;
            else if (mat == 1) kb [(((size_t)bi * NHEAD + head) * SEQ + s) * DK + dk] = v;
            else               vtb[(((size_t)bi * NHEAD + head) * DK + dk) * SEQ + s] = v;
        }
    }
}

// ---------------------------------------------------------------------------
// Kernel 2: attention. One workgroup (4 waves) per (b,h,16-query tile).
// Full 16x2048 score tile in LDS (f32) + bf16 probabilities (192KB of 320KB).
// ---------------------------------------------------------------------------
__global__ __launch_bounds__(128) void attn_kernel(
    const unsigned short* __restrict__ qb, const unsigned short* __restrict__ kb,
    const unsigned short* __restrict__ vtb, const int* __restrict__ mask,
    unsigned short* __restrict__ ctxb)
{
    __shared__ __align__(16) float          sc[16 * SEQ];     // 128 KB scores
    __shared__ __align__(16) unsigned short pb[16 * SEQ];     //  64 KB probs (bf16)
    __shared__ float rowsum[16];

    const int qtile = blockIdx.x & 127;
    const int bh    = blockIdx.x >> 7;
    const int b     = bh / NHEAD, h = bh - b * NHEAD;
    const int w     = threadIdx.x >> 5;
    const int lane  = threadIdx.x & 31;
    const int h4    = lane >> 4, nl = lane & 15;

    const unsigned short* qhead = qb  + (size_t)bh * SEQ * DK;
    const unsigned short* khead = kb  + (size_t)bh * SEQ * DK;
    const unsigned short* vhead = vtb + (size_t)bh * DK * SEQ;
    const int* mrow = mask + b * SEQ;

    // --- Phase 1: scores = Q K^T / sqrt(dk), masked, into LDS -------------
    Frag16 aq0, aq1;
    {
        const unsigned short* qr = qhead + (size_t)(qtile * 16 + nl) * DK;
        aq0.q[0] = *(const v4u*)(qr + 16 * h4);
        aq0.q[1] = *(const v4u*)(qr + 16 * h4 + 8);
        aq1.q[0] = *(const v4u*)(qr + 32 + 16 * h4);
        aq1.q[1] = *(const v4u*)(qr + 32 + 16 * h4 + 8);
    }
    auto loadK = [&](Frag16& f0, Frag16& f1, int j) {
        const unsigned short* kr = khead + (size_t)(j * 16 + nl) * DK;
        f0.q[0] = *(const v4u*)(kr + 16 * h4);
        f0.q[1] = *(const v4u*)(kr + 16 * h4 + 8);
        f1.q[0] = *(const v4u*)(kr + 32 + 16 * h4);
        f1.q[1] = *(const v4u*)(kr + 32 + 16 * h4 + 8);
    };
    auto scoreTile = [&](const Frag16& f0, const Frag16& f1, int j) {
        v8f c = {};
        c = wmma_bf16(aq0, f0, c);
        c = wmma_bf16(aq1, f1, c);
        const int col = j * 16 + nl;
        const int mk  = mrow[col];
        #pragma unroll
        for (int g = 0; g < 8; ++g) {
            const int m = g + 8 * h4;
            sc[m * SEQ + col] = mk ? c[g] * 0.125f : -__builtin_inff();
        }
    };

    Frag16 kA0, kA1, kB0, kB1;
    loadK(kA0, kA1, w);
    for (int i = 0; i < 32; i += 2) {        // 32 key tiles per wave, 2 per iter
        const int j0 = w + i * 4, j1 = w + (i + 1) * 4;
        loadK(kB0, kB1, j1);
        scoreTile(kA0, kA1, j0);
        const int j2 = (i + 2 < 32) ? w + (i + 2) * 4 : w;   // safe wrap on tail
        loadK(kA0, kA1, j2);
        scoreTile(kB0, kB1, j1);
    }
    __syncthreads();

    // --- Phase 2: softmax (8 lanes per row, shuffle reductions) -----------
    {
        const int row = threadIdx.x >> 3, sub = threadIdx.x & 7;
        float* srow = sc + row * SEQ;
        unsigned short* prow = pb + row * SEQ;
        float mx = -__builtin_inff();
        for (int i = sub * 256; i < sub * 256 + 256; ++i) mx = fmaxf(mx, srow[i]);
        #pragma unroll
        for (int o = 1; o < 8; o <<= 1) mx = fmaxf(mx, __shfl_xor(mx, o, 8));
        float sum = 0.f;
        for (int i = sub * 256; i < sub * 256 + 256; ++i) {
            const float p = __expf(srow[i] - mx);
            prow[i] = f2bf(p);
            sum += p;
        }
        #pragma unroll
        for (int o = 1; o < 8; o <<= 1) sum += __shfl_xor(sum, o, 8);
        if (sub == 0) rowsum[row] = sum;
    }
    __syncthreads();

    // --- Phase 3: ctx = P V (wave w owns 16 dk-columns) -------------------
    const unsigned short* vcol = vhead + (size_t)(w * 16 + nl) * SEQ;  // B col n
    const unsigned short* prow = pb + nl * SEQ;                        // A row m
    auto loadPV = [&](Frag16& a, Frag16& bf, int k0) {
        a.q[0]  = *(const v4u*)(prow + k0 + 16 * h4);
        a.q[1]  = *(const v4u*)(prow + k0 + 16 * h4 + 8);
        bf.q[0] = *(const v4u*)(vcol + k0 + 16 * h4);
        bf.q[1] = *(const v4u*)(vcol + k0 + 16 * h4 + 8);
    };

    v8f c = {};
    Frag16 pA, vA, pB, vB;
    loadPV(pA, vA, 0);
    for (int k0 = 0; k0 < SEQ; k0 += 64) {
        loadPV(pB, vB, k0 + 32);                        // k0+32 <= 2016, in range
        c = wmma_bf16(pA, vA, c);
        const int kn = (k0 + 64 < SEQ) ? k0 + 64 : 0;   // safe wrap on tail
        loadPV(pA, vA, kn);
        c = wmma_bf16(pB, vB, c);
    }
    #pragma unroll
    for (int g = 0; g < 8; ++g) {
        const int m   = g + 8 * h4;
        const float s = 1.0f / rowsum[m];
        const int token = b * SEQ + qtile * 16 + m;
        const int d     = h * DK + w * 16 + nl;
        ctxb[(size_t)token * DMODEL + d] = f2bf(c[g] * s);
    }
}

// ---------------------------------------------------------------------------
// Kernel 3: out = ctx @ Wo^T + bo (fp32 output).
// ---------------------------------------------------------------------------
__global__ __launch_bounds__(128) void proj_kernel(
    const unsigned short* __restrict__ ctxb, const unsigned short* __restrict__ wob,
    const float* __restrict__ bo, float* __restrict__ out)
{
    const int wid  = blockIdx.x * 4 + (threadIdx.x >> 5);
    const int lane = threadIdx.x & 31;
    const int h4   = lane >> 4, nl = lane & 15;
    const int rowTile  = wid / 12;
    const int colGroup = wid % 12;

    const unsigned short* arow  = ctxb + (size_t)(rowTile * 16 + nl) * DMODEL;
    const unsigned short* wbase = wob + (size_t)(colGroup * 64 + nl) * DMODEL + 16 * h4;

    auto loadA = [&](Frag16& a, int k0) {
        a.q[0] = *(const v4u*)(arow + k0 + 16 * h4);
        a.q[1] = *(const v4u*)(arow + k0 + 16 * h4 + 8);
    };
    auto loadB = [&](Frag16* bf, int k0) {
        #pragma unroll
        for (int t = 0; t < 4; ++t) {
            const unsigned short* wr = wbase + (size_t)t * 16 * DMODEL + k0;
            bf[t].q[0] = *(const v4u*)(wr);
            bf[t].q[1] = *(const v4u*)(wr + 8);
        }
    };

    v8f acc[4] = {};
    Frag16 a0, a1, b0[4], b1[4];
    loadA(a0, 0); loadB(b0, 0);

    for (int k0 = 0; k0 < DMODEL; k0 += 64) {
        loadA(a1, k0 + 32); loadB(b1, k0 + 32);
        #pragma unroll
        for (int t = 0; t < 4; ++t) acc[t] = wmma_bf16(a0, b0[t], acc[t]);
        const int kn = (k0 + 64 < DMODEL) ? k0 + 64 : 0;
        loadA(a0, kn); loadB(b0, kn);
        #pragma unroll
        for (int t = 0; t < 4; ++t) acc[t] = wmma_bf16(a1, b1[t], acc[t]);
    }

    #pragma unroll
    for (int t = 0; t < 4; ++t) {
        const int d  = colGroup * 64 + t * 16 + nl;
        const float bb = bo[d];
        #pragma unroll
        for (int g = 0; g < 8; ++g) {
            const int m = g + 8 * h4;
            const int r = rowTile * 16 + m;
            out[(size_t)r * DMODEL + d] = acc[t][g] + bb;
        }
    }
}

// ---------------------------------------------------------------------------
extern "C" void kernel_launch(void* const* d_in, const int* in_sizes, int n_in,
                              void* d_out, int out_size, void* d_ws, size_t ws_size,
                              hipStream_t stream) {
    const float* x    = (const float*)d_in[0];
    const int*   mask = (const int*)  d_in[1];
    const float* Wq   = (const float*)d_in[2];
    const float* bq   = (const float*)d_in[3];
    const float* Wk   = (const float*)d_in[4];
    const float* bk   = (const float*)d_in[5];
    const float* Wv   = (const float*)d_in[6];
    const float* bv   = (const float*)d_in[7];
    const float* Wo   = (const float*)d_in[8];
    const float* bo   = (const float*)d_in[9];
    float* out = (float*)d_out;

    char* ws = (char*)d_ws;
    const size_t segQKV = (size_t)2 * NHEAD * SEQ * DK * sizeof(unsigned short); // 6 MB
    const size_t segW   = (size_t)DMODEL * DMODEL * sizeof(unsigned short);      // 1.125 MB
    unsigned short* qb   = (unsigned short*)(ws);
    unsigned short* kb   = (unsigned short*)(ws + segQKV);
    unsigned short* vtb  = (unsigned short*)(ws + 2 * segQKV);
    unsigned short* ctxb = (unsigned short*)(ws + 3 * segQKV);   // [4096 x 768] bf16
    unsigned short* xbb  = (unsigned short*)(ws + 4 * segQKV);   // x in bf16 (6 MB)
    unsigned short* wqb  = (unsigned short*)(ws + 5 * segQKV);
    unsigned short* wkb  = (unsigned short*)(ws + 5 * segQKV + segW);
    unsigned short* wvb  = (unsigned short*)(ws + 5 * segQKV + 2 * segW);
    unsigned short* wob  = (unsigned short*)(ws + 5 * segQKV + 3 * segW);

    // one-shot fp32 -> bf16 pre-pass (memory bound, ~33 MB total)
    const int nX8 = (2 * SEQ * DMODEL) / 8;        // 393216
    const int nW8 = (DMODEL * DMODEL) / 8;         // 73728
    cvt_kernel<<<dim3((nX8 + 255) / 256), dim3(256), 0, stream>>>(x,  xbb, nX8);
    cvt_kernel<<<dim3((nW8 + 255) / 256), dim3(256), 0, stream>>>(Wq, wqb, nW8);
    cvt_kernel<<<dim3((nW8 + 255) / 256), dim3(256), 0, stream>>>(Wk, wkb, nW8);
    cvt_kernel<<<dim3((nW8 + 255) / 256), dim3(256), 0, stream>>>(Wv, wvb, nW8);
    cvt_kernel<<<dim3((nW8 + 255) / 256), dim3(256), 0, stream>>>(Wo, wob, nW8);

    // 3 mats * 256 row-tiles * 12 col-groups = 9216 waves / 4 per block
    qkv_kernel <<<dim3(2304), dim3(128), 0, stream>>>(xbb, wqb, bq, wkb, bk, wvb, bv, qb, kb, vtb);
    // 2*12*128 query tiles
    attn_kernel<<<dim3(3072), dim3(128), 0, stream>>>(qb, kb, vtb, mask, ctxb);
    // 256 row-tiles * 12 col-groups = 3072 waves / 4 per block
    proj_kernel<<<dim3(768),  dim3(128), 0, stream>>>(ctxb, wob, bo, out);
}
